// GCN_48172353191956
// MI455X (gfx1250) — matrix operand order repaired and checked by reading, not compiled, not run on hardware
//
#include <hip/hip_runtime.h>

#define NN      100000
#define EE      1600000
#define FI_ALL  128
#define HIDDEN  128
#define OUT_DIM 64

typedef __attribute__((ext_vector_type(16))) __bf16 v16bf;
typedef __attribute__((ext_vector_type(8)))  float  v8f;

// ---------------- degree / normalization ----------------

__global__ void deg_init_kernel(int* __restrict__ deg) {
  int i = blockIdx.x * blockDim.x + threadIdx.x;
  if (i < NN) deg[i] = 1;  // self-loop
}

__global__ void deg_count_kernel(const int* __restrict__ dst, int* __restrict__ deg) {
  int e = blockIdx.x * blockDim.x + threadIdx.x;
  if (e < EE) atomicAdd(&deg[dst[e]], 1);
}

__global__ void dinv_kernel(const int* __restrict__ deg, float* __restrict__ dinv) {
  int i = blockIdx.x * blockDim.x + threadIdx.x;
  if (i < NN) {
    int d = deg[i];
    dinv[i] = (d > 0) ? rsqrtf((float)d) : 0.0f;
  }
}

// ---------------- weight pre-swizzle into WMMA B-operand layout ----------------
// B (bf16 32x16): lanes 0-15 hold col N=lane, K=0..7,16..23 (2 per VGPR); lanes 16-31
// hold the same cols with K=8..15,24..31.  Stored so the GEMM reads one aligned v16bf
// (32B) per (n-tile, k-chunk, lane):  Wsw[((nt*KC + kc)*32 + lane)*16 + 0..15]
template <int FO>
__global__ void wswz_kernel(const float* __restrict__ W, __bf16* __restrict__ Wsw) {
  constexpr int KC = FI_ALL / 32;
  constexpr int NT = FO / 16;
  int t = blockIdx.x * blockDim.x + threadIdx.x;
  if (t >= NT * KC * 32) return;
  int lane = t & 31;
  int kc   = (t >> 5) % KC;
  int nt   = (t >> 5) / KC;
  int col   = nt * 16 + (lane & 15);
  int kbase = kc * 32 + ((lane >> 4) << 3);
  __bf16* __restrict__ op = Wsw + (size_t)t * 16;
#pragma unroll
  for (int j = 0; j < 8; ++j) {
    op[2 * j]     = (__bf16)W[(size_t)(kbase + 2 * j + (j >= 4 ? 8 : 0)) * FO + col];
    op[2 * j + 1] = (__bf16)W[(size_t)(kbase + 2 * j + 1 + (j >= 4 ? 8 : 0)) * FO + col];
  }
}

// ---------------- dense GEMM: Out[N,FO] = reluopt(A[N,128]) @ W ----------------
// One wave per 16-row strip: holds FO/16 accumulators, loads each A chunk once,
// fully unrolled -> 4*(FO/16) v_wmma_f32_16x16x32_bf16 with immediate offsets.
__device__ __forceinline__ float4 relu4(float4 v) {
  return make_float4(fmaxf(v.x, 0.f), fmaxf(v.y, 0.f), fmaxf(v.z, 0.f), fmaxf(v.w, 0.f));
}

template <int FO>
__global__ void gcn_gemm_wmma_kernel(const float* __restrict__ A,
                                     const __bf16* __restrict__ Wsw,
                                     float* __restrict__ Out, int relu_in) {
  constexpr int KC = FI_ALL / 32;   // 4 k-chunks
  constexpr int NT = FO / 16;       // n-tiles held per wave
  const int lane = threadIdx.x & 31;
  const int wib  = threadIdx.x >> 5;
  const int mt   = blockIdx.x * (blockDim.x >> 5) + wib;   // wave-uniform
  if (mt >= NN / 16) return;

  const int m0    = mt * 16;
  const int rowA  = m0 + (lane & 15);
  const int koffA = (lane >> 4) * 8;
  const float* __restrict__ arow = A + (size_t)rowA * FI_ALL + koffA;

  v8f acc[NT] = {};

#pragma unroll
  for (int kc = 0; kc < KC; ++kc) {
    // A layout (bf16 16x32): lane half selects K base 0/8; vector elems 0..7 are
    // K=+0..7, elems 8..15 are K=+16..23 (relative to lane's base).
    float4 f0 = *(const float4*)(arow + kc * 32);
    float4 f1 = *(const float4*)(arow + kc * 32 + 4);
    float4 f2 = *(const float4*)(arow + kc * 32 + 16);
    float4 f3 = *(const float4*)(arow + kc * 32 + 20);
    if (relu_in) { f0 = relu4(f0); f1 = relu4(f1); f2 = relu4(f2); f3 = relu4(f3); }
    v16bf a;
    a[0]  = (__bf16)f0.x; a[1]  = (__bf16)f0.y; a[2]  = (__bf16)f0.z; a[3]  = (__bf16)f0.w;
    a[4]  = (__bf16)f1.x; a[5]  = (__bf16)f1.y; a[6]  = (__bf16)f1.z; a[7]  = (__bf16)f1.w;
    a[8]  = (__bf16)f2.x; a[9]  = (__bf16)f2.y; a[10] = (__bf16)f2.z; a[11] = (__bf16)f2.w;
    a[12] = (__bf16)f3.x; a[13] = (__bf16)f3.y; a[14] = (__bf16)f3.z; a[15] = (__bf16)f3.w;

#pragma unroll
    for (int nt = 0; nt < NT; ++nt) {
      v16bf b = *(const v16bf*)(Wsw + ((size_t)(nt * KC + kc) * 32 + lane) * 16);
      acc[nt] = __builtin_amdgcn_wmma_f32_16x16x32_bf16(false, a, false, b,
                                                        (short)0, acc[nt],
                                                        false, false);
    }
  }

  // C/D layout: lane&15 = column, VGPR v -> row = v + 8*(lane>>4)
  const int rbase = m0 + ((lane >> 4) << 3);
  float* __restrict__ obase = Out + (size_t)rbase * FO + (lane & 15);
#pragma unroll
  for (int nt = 0; nt < NT; ++nt)
#pragma unroll
    for (int v = 0; v < 8; ++v)
      obase[(size_t)v * FO + nt * 16] = acc[nt][v];
}

// ---------------- aggregation init: agg = bias + h * dinv^2 (fused self-loop) -------

template <int FO>
__global__ void agg_init_kernel(float* __restrict__ agg,
                                const float* __restrict__ bias,
                                const float* __restrict__ h,
                                const float* __restrict__ dinv) {
  constexpr int SH = (FO == 128) ? 7 : 6;
  long long i = (long long)blockIdx.x * blockDim.x + threadIdx.x;
  if (i >= (long long)NN * FO) return;
  int node = (int)(i >> SH);
  int f    = (int)(i & (FO - 1));
  float di = dinv[node];
  agg[i] = bias[f] + h[i] * di * di;
}

// ---------------- edge scatter: one wave per edge ----------------

template <int FO>
__global__ void edge_scatter_kernel(const int* __restrict__ src,
                                    const int* __restrict__ dst,
                                    const float* __restrict__ dinv,
                                    const float* __restrict__ h,
                                    float* __restrict__ agg) {
  constexpr int C = FO / 32;  // floats per lane
  long long gt = (long long)blockIdx.x * blockDim.x + threadIdx.x;
  long long e  = gt >> 5;
  int lane     = (int)(gt & 31);
  if (e >= EE) return;

  int s = src[e];
  int d = dst[e];
  float nrm = dinv[s] * dinv[d];

  const float* __restrict__ hp = h   + (size_t)s * FO + lane * C;
  float*       __restrict__ op = agg + (size_t)d * FO + lane * C;
  if constexpr (C == 4) {
    float4 v = *(const float4*)hp;                 // coalesced 512B/wave row read
    unsafeAtomicAdd(op + 0, v.x * nrm);            // global_atomic_add_f32
    unsafeAtomicAdd(op + 1, v.y * nrm);
    unsafeAtomicAdd(op + 2, v.z * nrm);
    unsafeAtomicAdd(op + 3, v.w * nrm);
  } else {
    float2 v = *(const float2*)hp;
    unsafeAtomicAdd(op + 0, v.x * nrm);
    unsafeAtomicAdd(op + 1, v.y * nrm);
  }
}

__global__ void relu_kernel(float* __restrict__ x, long long n) {
  long long i = (long long)blockIdx.x * blockDim.x + threadIdx.x;
  if (i < n) x[i] = fmaxf(x[i], 0.0f);
}

// ---------------- host side ----------------

extern "C" void kernel_launch(void* const* d_in, const int* in_sizes, int n_in,
                              void* d_out, int out_size, void* d_ws, size_t ws_size,
                              hipStream_t stream) {
  (void)in_sizes; (void)n_in; (void)out_size; (void)ws_size;

  const float* x  = (const float*)d_in[0];
  const int*   ei = (const int*)d_in[1];           // [2, EE]: row0=src, row1=dst
  const float* Wl[4] = {(const float*)d_in[2], (const float*)d_in[4],
                        (const float*)d_in[6], (const float*)d_in[8]};
  const float* Bl[4] = {(const float*)d_in[3], (const float*)d_in[5],
                        (const float*)d_in[7], (const float*)d_in[9]};
  const int* src = ei;
  const int* dst = ei + EE;

  char* ws = (char*)d_ws;
  int*    deg  = (int*)(ws);                               // N ints
  float*  dinv = (float*)(ws + (size_t)512 * 1024);        // N floats
  __bf16* wsw  = (__bf16*)(ws + (size_t)768 * 1024);       // 32KB swizzled W (reused)
  float*  bufH = (float*)(ws + (size_t)1024 * 1024);       // N*128 fp32
  float*  bufT = (float*)(ws + (size_t)1024 * 1024 +
                               (size_t)NN * HIDDEN * sizeof(float));

  deg_init_kernel<<<(NN + 255) / 256, 256, 0, stream>>>(deg);
  deg_count_kernel<<<(EE + 255) / 256, 256, 0, stream>>>(dst, deg);
  dinv_kernel<<<(NN + 255) / 256, 256, 0, stream>>>(deg, dinv);

  const int mtiles = NN / 16;                       // 6250, exact
  const int gemm_blocks = (mtiles + 7) / 8;         // 8 waves / 256-thread block

  const float* hin = x;
  for (int l = 0; l < 4; ++l) {
    const bool last = (l == 3);
    float* agg = last ? (float*)d_out : bufH;
    const int relu_in = (l > 0) ? 1 : 0;

    if (!last) {
      wswz_kernel<HIDDEN><<<4, 256, 0, stream>>>(Wl[l], wsw);
      gcn_gemm_wmma_kernel<HIDDEN><<<gemm_blocks, 256, 0, stream>>>(hin, wsw, bufT, relu_in);
      long long tot = (long long)NN * HIDDEN;
      agg_init_kernel<HIDDEN><<<(unsigned)((tot + 255) / 256), 256, 0, stream>>>(
          agg, Bl[l], bufT, dinv);
      long long thr = (long long)EE * 32;
      edge_scatter_kernel<HIDDEN><<<(unsigned)((thr + 255) / 256), 256, 0, stream>>>(
          src, dst, dinv, bufT, agg);
    } else {
      wswz_kernel<OUT_DIM><<<2, 256, 0, stream>>>(Wl[l], wsw);
      gcn_gemm_wmma_kernel<OUT_DIM><<<gemm_blocks, 256, 0, stream>>>(hin, wsw, bufT, relu_in);
      long long tot = (long long)NN * OUT_DIM;
      agg_init_kernel<OUT_DIM><<<(unsigned)((tot + 255) / 256), 256, 0, stream>>>(
          agg, Bl[l], bufT, dinv);
      long long thr = (long long)EE * 32;
      edge_scatter_kernel<OUT_DIM><<<(unsigned)((thr + 255) / 256), 256, 0, stream>>>(
          src, dst, dinv, bufT, agg);
    }
    hin = bufH;                                  // pre-ReLU h; ReLU fused next layer
  }

  long long on = (long long)NN * OUT_DIM;
  relu_kernel<<<(unsigned)((on + 255) / 256), 256, 0, stream>>>((float*)d_out, on);
}
